// BERT_66898410602566
// MI455X (gfx1250) — matrix operand order repaired and checked by reading, not compiled, not run on hardware
//
#include <hip/hip_runtime.h>
#include <math.h>

#define BB 16
#define SS 512
#define DD 1024
#define HH 16
#define LL 12
#define DFF 4096
#define DK 64

typedef __attribute__((ext_vector_type(16))) __bf16 v16bf;
typedef __attribute__((ext_vector_type(8)))  float  v8f;
typedef __attribute__((ext_vector_type(4)))  unsigned v4u;
typedef __attribute__((ext_vector_type(8)))  int      v8i_t;
typedef __attribute__((ext_vector_type(4)))  int      v4i_t;
typedef unsigned short ushort_t;

union FragB {
    v16bf v;
    unsigned u[8];
};

__device__ inline ushort_t f2bf(float f) {
    unsigned u = __builtin_bit_cast(unsigned, f);
    unsigned rounding = 0x7FFFu + ((u >> 16) & 1u);
    return (ushort_t)((u + rounding) >> 16);
}

__device__ inline v8f v8f_zero() {
    v8f z = {0.f, 0.f, 0.f, 0.f, 0.f, 0.f, 0.f, 0.f};
    return z;
}

// ---------------------------------------------------------------------------
// TDM: 2D tile (bf16) global -> LDS.  D# built per CDNA5 ISA 08_async_tensor:
// group0: count=1 | lds_addr[63:32] | global_addr[120:64] | type=2[127:126]
// group1: data_size=1(2B) | tensor_dim0[79:48] | tensor_dim1[111:80]
//         | tile_dim0[127:112] | tile_dim1[143:128] | dim0_stride[207:160]
// This toolchain's builtin is the 6-arg form (g0, g1, g2, g3, g4, cpol).
// ---------------------------------------------------------------------------
__device__ inline void tdm_load_2d(unsigned lds_addr, const void* gptr,
                                   unsigned tile_k, unsigned tile_rows,
                                   unsigned tensor_k, unsigned tensor_rows,
                                   unsigned long long stride_k /*elements*/) {
    unsigned long long ga = (unsigned long long)gptr;
    v4u g0;
    g0[0] = 1u;                                             // count=1
    g0[1] = lds_addr;                                       // lds byte addr
    g0[2] = (unsigned)(ga & 0xffffffffu);                   // ga[31:0]
    g0[3] = (unsigned)((ga >> 32) & 0x01ffffffu) | (2u << 30); // ga[56:32]|type=2
    v8i_t g1;
    g1[0] = (int)(1u << 16);                                // data_size = 2 bytes
    g1[1] = (int)((tensor_k & 0xffffu) << 16);              // dim0[15:0]
    g1[2] = (int)((tensor_k >> 16) | ((tensor_rows & 0xffffu) << 16));
    g1[3] = (int)((tensor_rows >> 16) | (tile_k << 16));    // dim1 hi | tile_dim0
    g1[4] = (int)(tile_rows & 0xffffu);                     // tile_dim1 (tile_dim2=0)
    g1[5] = (int)(stride_k & 0xffffffffu);                  // dim0_stride[31:0]
    g1[6] = (int)((stride_k >> 32) & 0xffffu);              // dim0_stride[47:32]
    g1[7] = 0;
    v4i_t z4 = {0, 0, 0, 0};
    v8i_t z8 = {0, 0, 0, 0, 0, 0, 0, 0};
    __builtin_amdgcn_tensor_load_to_lds(g0, g1, z4, z4, z8, 0);
}

// ---------------------------------------------------------------------------
// Weight convert+transpose: W[K,N] fp32 -> Wt[N,K] bf16 (blockIdx.z = layer)
// ---------------------------------------------------------------------------
__global__ __launch_bounds__(256)
void convt_kernel(const float* __restrict__ W, ushort_t* __restrict__ Wt,
                  int K, int N) {
    __shared__ float tile[32][33];
    const float* Wl = W + (size_t)blockIdx.z * K * N;
    ushort_t* Wtl = Wt + (size_t)blockIdx.z * K * N;
    int n0 = blockIdx.x * 32, k0 = blockIdx.y * 32;
    int tx = threadIdx.x & 31, ty = threadIdx.x >> 5;   // 32 x 8
#pragma unroll
    for (int i = 0; i < 4; ++i) {
        int k = ty + i * 8;
        tile[k][tx] = Wl[(size_t)(k0 + k) * N + n0 + tx];
    }
    __syncthreads();
#pragma unroll
    for (int i = 0; i < 4; ++i) {
        int n = ty + i * 8;
        Wtl[(size_t)(n0 + n) * K + k0 + tx] = f2bf(tile[tx][n]);
    }
}

// ---------------------------------------------------------------------------
// Embedding: x = tok_emb[masked] + PE + seg_emb[1]   (fp32, vectorized)
// ---------------------------------------------------------------------------
__global__ __launch_bounds__(256)
void embed_kernel(const int* __restrict__ masked, const float* __restrict__ tok,
                  const float* __restrict__ seg, float* __restrict__ x) {
    int row = blockIdx.x;
    int s = row & (SS - 1);
    int tid = threadIdx.x;
    int tokid = masked[row];
    const float negln = -logf(10000.f) / (float)DD;
    int d0 = tid * 4;
    float4 tv = ((const float4*)(tok + (size_t)tokid * DD))[tid];
    float4 sv = ((const float4*)(seg + DD))[tid];
    float a0 = (float)s * __expf(negln * (float)d0);
    float a1 = (float)s * __expf(negln * (float)(d0 + 2));
    float4 r;
    r.x = tv.x + __sinf(a0) + sv.x;
    r.y = tv.y + __cosf(a0) + sv.y;
    r.z = tv.z + __sinf(a1) + sv.z;
    r.w = tv.w + __cosf(a1) + sv.w;
    ((float4*)(x + (size_t)row * DD))[tid] = r;
}

// ---------------------------------------------------------------------------
// LayerNorm: fp32 in -> bf16 out, one block per row, vectorized
// ---------------------------------------------------------------------------
__global__ __launch_bounds__(256)
void layernorm_kernel(const float* __restrict__ x, const float* __restrict__ scale,
                      const float* __restrict__ bias, ushort_t* __restrict__ y) {
    __shared__ float red[256];
    int row = blockIdx.x, tid = threadIdx.x;
    float4 xv = ((const float4*)(x + (size_t)row * DD))[tid];
    float s = xv.x + xv.y + xv.z + xv.w;
    float s2 = xv.x * xv.x + xv.y * xv.y + xv.z * xv.z + xv.w * xv.w;
    red[tid] = s; __syncthreads();
    for (int d = 128; d > 0; d >>= 1) {
        if (tid < d) red[tid] += red[tid + d];
        __syncthreads();
    }
    float mean = red[0] * (1.f / DD);
    __syncthreads();
    red[tid] = s2; __syncthreads();
    for (int d = 128; d > 0; d >>= 1) {
        if (tid < d) red[tid] += red[tid + d];
        __syncthreads();
    }
    float var = red[0] * (1.f / DD) - mean * mean;
    float inv = rsqrtf(var + 1e-5f);
    float4 sc = ((const float4*)scale)[tid];
    float4 bi = ((const float4*)bias)[tid];
    float o0 = (xv.x - mean) * inv * sc.x + bi.x;
    float o1 = (xv.y - mean) * inv * sc.y + bi.y;
    float o2 = (xv.z - mean) * inv * sc.z + bi.z;
    float o3 = (xv.w - mean) * inv * sc.w + bi.w;
    uint2 p;
    p.x = (unsigned)f2bf(o0) | ((unsigned)f2bf(o1) << 16);
    p.y = (unsigned)f2bf(o2) | ((unsigned)f2bf(o3) << 16);
    ((uint2*)(y + (size_t)row * DD))[tid] = p;
}

// ---------------------------------------------------------------------------
// bf16 WMMA GEMM with TDM-staged, double-buffered LDS tiles.
// A[M,K] bf16 row-major, Bt[N,K] bf16 (pre-transposed weights).
// Block tile 128x256, BK=32, 8 waves (2x4), wave tile 64x64 -> 16 WMMA/iter.
// Epilogue: +bias [+res fp32] [gelu], out fp32 or bf16.
// ---------------------------------------------------------------------------
#define BM 128
#define BN 256
#define BK 32

__device__ inline float gelu_exact(float v) {
    return 0.5f * v * (1.f + erff(v * 0.70710678118654752f));
}

template <bool GELU_ACT, bool RES, bool OUTBF>
__global__ __launch_bounds__(256)
void gemm_kernel(const ushort_t* __restrict__ A, const ushort_t* __restrict__ Bt,
                 const float* __restrict__ bias, const float* __restrict__ res,
                 void* __restrict__ Cout, int M, int N, int K) {
    __shared__ ushort_t As[2][BM * BK];
    __shared__ ushort_t Bs[2][BN * BK];
    int tid = threadIdx.x;
    int lane = tid & 31, wave = tid >> 5;
    int wm = wave & 1, wn = wave >> 1;           // 2 x 4 wave grid
    int m0 = blockIdx.y * BM, n0 = blockIdx.x * BN;
    int half = lane >> 4, l16 = lane & 15;

    v8f acc[4][4];
#pragma unroll
    for (int mi = 0; mi < 4; ++mi)
#pragma unroll
        for (int ni = 0; ni < 4; ++ni) acc[mi][ni] = v8f_zero();

    int ntiles = K / BK;
    if (wave == 0) {
        tdm_load_2d((unsigned)(size_t)&As[0][0], A + (size_t)m0 * K,
                    BK, BM, (unsigned)K, (unsigned)M, (unsigned long long)K);
        tdm_load_2d((unsigned)(size_t)&Bs[0][0], Bt + (size_t)n0 * K,
                    BK, BN, (unsigned)K, (unsigned)N, (unsigned long long)K);
    }

    for (int it = 0; it < ntiles; ++it) {
        int cur = it & 1;
        if (wave == 0) {
            if (it + 1 < ntiles) {
                size_t k0 = (size_t)(it + 1) * BK;
                tdm_load_2d((unsigned)(size_t)&As[cur ^ 1][0], A + (size_t)m0 * K + k0,
                            BK, BM, (unsigned)K, (unsigned)M, (unsigned long long)K);
                tdm_load_2d((unsigned)(size_t)&Bs[cur ^ 1][0], Bt + (size_t)n0 * K + k0,
                            BK, BN, (unsigned)K, (unsigned)N, (unsigned long long)K);
                __builtin_amdgcn_s_wait_tensorcnt(2);  // current tile landed
            } else {
                __builtin_amdgcn_s_wait_tensorcnt(0);
            }
        }
        __syncthreads();

        const unsigned* As32 = (const unsigned*)&As[cur][0];
        const unsigned* Bs32 = (const unsigned*)&Bs[cur][0];
        FragB a[4], b[4];
#pragma unroll
        for (int mi = 0; mi < 4; ++mi) {
            int m = wm * 64 + mi * 16 + l16;
#pragma unroll
            for (int j = 0; j < 8; ++j) {
                int k = (j < 4) ? (2 * j + 8 * half) : (16 + 2 * (j - 4) + 8 * half);
                a[mi].u[j] = As32[(m * BK + k) >> 1];
            }
        }
#pragma unroll
        for (int ni = 0; ni < 4; ++ni) {
            int n = wn * 64 + ni * 16 + l16;
#pragma unroll
            for (int j = 0; j < 8; ++j) {
                int k = (j < 4) ? (2 * j + 8 * half) : (16 + 2 * (j - 4) + 8 * half);
                b[ni].u[j] = Bs32[(n * BK + k) >> 1];
            }
        }
#pragma unroll
        for (int mi = 0; mi < 4; ++mi)
#pragma unroll
            for (int ni = 0; ni < 4; ++ni)
                acc[mi][ni] = __builtin_amdgcn_wmma_f32_16x16x32_bf16(
                    false, a[mi].v, false, b[ni].v, (short)0, acc[mi][ni], false, false);
        __syncthreads();
    }

#pragma unroll
    for (int mi = 0; mi < 4; ++mi)
#pragma unroll
        for (int ni = 0; ni < 4; ++ni) {
            int n = n0 + wn * 64 + ni * 16 + l16;
            float bv = bias[n];
#pragma unroll
            for (int r = 0; r < 8; ++r) {
                int m = m0 + wm * 64 + mi * 16 + r + 8 * half;
                float v = acc[mi][ni][r] + bv;
                if (RES) v += res[(size_t)m * N + n];
                if (GELU_ACT) v = gelu_exact(v);
                if (OUTBF) ((ushort_t*)Cout)[(size_t)m * N + n] = f2bf(v);
                else       ((float*)Cout)[(size_t)m * N + n] = v;
            }
        }
}

// ---------------------------------------------------------------------------
// Fused flash attention over bf16 t[B,S,H*DK]; out bf16.
// Block = (b,h, 128-query chunk), 8 waves x 16 queries, 32-key LDS tiles.
// ---------------------------------------------------------------------------
__global__ __launch_bounds__(256)
void attn_kernel(const ushort_t* __restrict__ t, const int* __restrict__ masked,
                 ushort_t* __restrict__ o) {
    __shared__ ushort_t Ks[32 * 64];       // [key][dk]
    __shared__ ushort_t VT[64 * 32];       // [dk][key]
    __shared__ ushort_t Ps[8 * 16 * 32];   // per-wave P tile [m][k]

    int tid = threadIdx.x, lane = tid & 31, wave = tid >> 5;
    int bh = blockIdx.y;
    int b = bh >> 4, h = bh & 15;
    int q_base = blockIdx.x * 128 + wave * 16;
    int half = lane >> 4, l16 = lane & 15;
    const float scale = 0.125f;

    // Q fragments straight from bf16 global (u32 pair loads)
    FragB qf[2];
    {
        int q = q_base + l16;
        const ushort_t* qp = t + ((size_t)(b * SS + q)) * DD + h * DK;
#pragma unroll
        for (int f = 0; f < 2; ++f)
#pragma unroll
            for (int j = 0; j < 8; ++j) {
                int dk = 32 * f +
                         ((j < 4) ? (2 * j + 8 * half) : (16 + 2 * (j - 4) + 8 * half));
                qf[f].u[j] = *(const unsigned*)(qp + dk);
            }
    }

    v8f ov[4];
#pragma unroll
    for (int nt = 0; nt < 4; ++nt) ov[nt] = v8f_zero();
    float rowmax[8], rowsum[8];
#pragma unroll
    for (int r = 0; r < 8; ++r) { rowmax[r] = -1e30f; rowsum[r] = 0.f; }

    ushort_t* Pw = Ps + wave * 16 * 32;
    const unsigned* Ks32 = (const unsigned*)Ks;
    const unsigned* VT32 = (const unsigned*)VT;
    const unsigned* Pw32 = (const unsigned*)Pw;

    int key_s = tid >> 3;               // 0..31
    int dkc = (tid & 7) * 8;            // 0..56 step 8

    for (int kb = 0; kb < SS / 32; ++kb) {
        int key0 = kb * 32;
        __syncthreads();
        {   // stage K (row-major) and V (transposed) -- pure bf16 copy
            uint4 kv = *(const uint4*)(t + ((size_t)(b * SS + key0 + key_s)) * DD +
                                       h * DK + dkc);
            *(uint4*)(Ks + key_s * 64 + dkc) = kv;
            VT[(dkc + 0) * 32 + key_s] = (ushort_t)kv.x;
            VT[(dkc + 1) * 32 + key_s] = (ushort_t)(kv.x >> 16);
            VT[(dkc + 2) * 32 + key_s] = (ushort_t)kv.y;
            VT[(dkc + 3) * 32 + key_s] = (ushort_t)(kv.y >> 16);
            VT[(dkc + 4) * 32 + key_s] = (ushort_t)kv.z;
            VT[(dkc + 5) * 32 + key_s] = (ushort_t)(kv.z >> 16);
            VT[(dkc + 6) * 32 + key_s] = (ushort_t)kv.w;
            VT[(dkc + 7) * 32 + key_s] = (ushort_t)(kv.w >> 16);
        }
        __syncthreads();

        v8f sc[2];
#pragma unroll
        for (int nt = 0; nt < 2; ++nt) {
            sc[nt] = v8f_zero();
            FragB kf[2];
            int key = nt * 16 + l16;
#pragma unroll
            for (int f = 0; f < 2; ++f)
#pragma unroll
                for (int j = 0; j < 8; ++j) {
                    int dk = 32 * f + ((j < 4) ? (2 * j + 8 * half) : (16 + 2 * (j - 4) + 8 * half));
                    kf[f].u[j] = Ks32[(key * 64 + dk) >> 1];
                }
            sc[nt] = __builtin_amdgcn_wmma_f32_16x16x32_bf16(
                false, qf[0].v, false, kf[0].v, (short)0, sc[nt], false, false);
            sc[nt] = __builtin_amdgcn_wmma_f32_16x16x32_bf16(
                false, qf[1].v, false, kf[1].v, (short)0, sc[nt], false, false);
        }

        float mb[2];
#pragma unroll
        for (int nt = 0; nt < 2; ++nt) {
            int key = key0 + nt * 16 + l16;
            mb[nt] = (masked[b * SS + key] != 1) ? 0.f : -1e9f;
        }

#pragma unroll
        for (int r = 0; r < 8; ++r) {
            float s0 = sc[0][r] * scale + mb[0];
            float s1 = sc[1][r] * scale + mb[1];
            float mx = fmaxf(s0, s1);
#pragma unroll
            for (int d = 1; d < 16; d <<= 1) mx = fmaxf(mx, __shfl_xor(mx, d, 32));
            float newmax = fmaxf(rowmax[r], mx);
            float corr = __expf(rowmax[r] - newmax);
            float p0 = __expf(s0 - newmax);
            float p1 = __expf(s1 - newmax);
            float ps = p0 + p1;
#pragma unroll
            for (int d = 1; d < 16; d <<= 1) ps += __shfl_xor(ps, d, 32);
            rowsum[r] = rowsum[r] * corr + ps;
            rowmax[r] = newmax;
#pragma unroll
            for (int nt = 0; nt < 4; ++nt) ov[nt][r] *= corr;
            int m = r + 8 * half;
            Pw[m * 32 + l16] = f2bf(p0);
            Pw[m * 32 + 16 + l16] = f2bf(p1);
        }
        __syncthreads();

        FragB pf;
#pragma unroll
        for (int j = 0; j < 8; ++j) {
            int k = (j < 4) ? (2 * j + 8 * half) : (16 + 2 * (j - 4) + 8 * half);
            pf.u[j] = Pw32[(l16 * 32 + k) >> 1];
        }
#pragma unroll
        for (int nt = 0; nt < 4; ++nt) {
            FragB vf;
            int dk = nt * 16 + l16;
#pragma unroll
            for (int j = 0; j < 8; ++j) {
                int k = (j < 4) ? (2 * j + 8 * half) : (16 + 2 * (j - 4) + 8 * half);
                vf.u[j] = VT32[(dk * 32 + k) >> 1];
            }
            ov[nt] = __builtin_amdgcn_wmma_f32_16x16x32_bf16(
                false, pf.v, false, vf.v, (short)0, ov[nt], false, false);
        }
    }

#pragma unroll
    for (int nt = 0; nt < 4; ++nt)
#pragma unroll
        for (int r = 0; r < 8; ++r) {
            int m = r + 8 * half;
            int q = q_base + m;
            int dk = nt * 16 + l16;
            o[((size_t)(b * SS + q)) * DD + h * DK + dk] = f2bf(ov[nt][r] / rowsum[r]);
        }
}

// ---------------------------------------------------------------------------
// Host launcher
// ---------------------------------------------------------------------------
extern "C" void kernel_launch(void* const* d_in, const int* in_sizes, int n_in,
                              void* d_out, int out_size, void* d_ws, size_t ws_size,
                              hipStream_t stream) {
    const int*   masked    = (const int*)  d_in[0];
    const float* tok_emb   = (const float*)d_in[1];
    const float* seg_emb   = (const float*)d_in[2];
    const float* ln1_scale = (const float*)d_in[3];
    const float* ln1_bias  = (const float*)d_in[4];
    const float* wq        = (const float*)d_in[5];
    const float* bq        = (const float*)d_in[6];
    const float* wo        = (const float*)d_in[7];
    const float* bo        = (const float*)d_in[8];
    const float* ln2_scale = (const float*)d_in[9];
    const float* ln2_bias  = (const float*)d_in[10];
    const float* w1        = (const float*)d_in[11];
    const float* b1        = (const float*)d_in[12];
    const float* w2        = (const float*)d_in[13];
    const float* b2        = (const float*)d_in[14];

    float* x = (float*)d_out;                  // running activation [B,S,D] fp32
    char* ws = (char*)d_ws;
    const size_t MB = 1024 * 1024;
    ushort_t* wqT = (ushort_t*)(ws);                 //  24 MB [L][D,D]^T
    ushort_t* woT = (ushort_t*)(ws + 24 * MB);       //  24 MB
    ushort_t* w1T = (ushort_t*)(ws + 48 * MB);       //  96 MB [L][DFF,D]
    ushort_t* w2T = (ushort_t*)(ws + 144 * MB);      //  96 MB [L][D,DFF]
    ushort_t* xnb = (ushort_t*)(ws + 240 * MB);      //  16 MB
    ushort_t* tb  = (ushort_t*)(ws + 256 * MB);      //  16 MB
    ushort_t* ob  = (ushort_t*)(ws + 272 * MB);      //  16 MB
    ushort_t* hb  = (ushort_t*)(ws + 288 * MB);      //  64 MB

    const int ROWS = BB * SS;                  // 8192
    dim3 blk(256);

    // One-time (per call) weight convert + transpose to bf16 [N,K]
    convt_kernel<<<dim3(DD / 32, DD / 32, LL), blk, 0, stream>>>(wq, wqT, DD, DD);
    convt_kernel<<<dim3(DD / 32, DD / 32, LL), blk, 0, stream>>>(wo, woT, DD, DD);
    convt_kernel<<<dim3(DFF / 32, DD / 32, LL), blk, 0, stream>>>(w1, w1T, DD, DFF);
    convt_kernel<<<dim3(DD / 32, DFF / 32, LL), blk, 0, stream>>>(w2, w2T, DFF, DD);

    embed_kernel<<<ROWS, blk, 0, stream>>>(masked, tok_emb, seg_emb, x);

    dim3 gProj(DD / BN, ROWS / BM);            // (4, 64)
    dim3 gFF1(DFF / BN, ROWS / BM);            // (16, 64)
    dim3 gAttn(SS / 128, BB * HH);             // (4, 256)

    for (int l = 0; l < LL; ++l) {
        layernorm_kernel<<<ROWS, blk, 0, stream>>>(x, ln1_scale + l * DD,
                                                   ln1_bias + l * DD, xnb);
        gemm_kernel<false, false, true><<<gProj, blk, 0, stream>>>(
            xnb, wqT + (size_t)l * DD * DD, bq + l * DD, nullptr, tb, ROWS, DD, DD);
        attn_kernel<<<gAttn, blk, 0, stream>>>(tb, masked, ob);
        gemm_kernel<false, true, false><<<gProj, blk, 0, stream>>>(
            ob, woT + (size_t)l * DD * DD, bo + l * DD, x, x, ROWS, DD, DD);
        layernorm_kernel<<<ROWS, blk, 0, stream>>>(x, ln2_scale + l * DD,
                                                   ln2_bias + l * DD, xnb);
        gemm_kernel<true, false, true><<<gFF1, blk, 0, stream>>>(
            xnb, w1T + (size_t)l * DD * DFF, b1 + l * DFF, nullptr, hb, ROWS, DFF, DD);
        gemm_kernel<false, true, false><<<gProj, blk, 0, stream>>>(
            hb, w2T + (size_t)l * DFF * DD, b2 + l * DD, x, x, ROWS, DD, DFF);
    }
}
